// FeatureDecodingLayer_438086664766
// MI455X (gfx1250) — matrix operand order repaired and checked by reading, not compiled, not run on hardware
//
#include <hip/hip_runtime.h>
#include <hip/hip_bf16.h>
#include <stdint.h>

typedef _Float16 f16;
typedef _Float16 v16h __attribute__((ext_vector_type(16)));
typedef _Float16 v8h  __attribute__((ext_vector_type(8)));
typedef float    v8f  __attribute__((ext_vector_type(8)));

#define B_    4
#define N1_   4096
#define N2_   1024
#define C1_   256
#define C2_   256
#define K_    16
#define WN_   16
#define MLP0_ 256
#define MLP1_ 256
#define FLAT_ (WN_*MLP0_ + C1_)   // 4352

// eval-mode BN (mean 0, var 1) + ReLU:  relu(x * g/sqrt(1+eps) + be)
__device__ __forceinline__ float bnrelu(float s, float g, float be) {
    const float invs = 0.9999950000374997f; // 1/sqrt(1 + 1e-5)
    return fmaxf(s * (g * invs) + be, 0.0f);
}

// ---------------------------------------------------------------------------
// f32 -> f16 conversion (weights / small tensors)
// ---------------------------------------------------------------------------
__global__ void cvt_f32_f16_kernel(const float* __restrict__ src,
                                   f16* __restrict__ dst, int n) {
    int t = blockIdx.x * 256 + threadIdx.x;
    if (t < n) dst[t] = (f16)src[t];
}

// ---------------------------------------------------------------------------
// three_nn + inverse-distance interpolation of feat_points2
// one wave32 per query point; xyz2 staged in LDS
// interp layout: [B][N1][C2] f32
// ---------------------------------------------------------------------------
__global__ __launch_bounds__(256) void three_nn_interp_kernel(
    const float* __restrict__ xyz1, const float* __restrict__ xyz2,
    const float* __restrict__ f2, float* __restrict__ interp)
{
    __shared__ float s2x[N2_], s2y[N2_], s2z[N2_];
    __shared__ float ld[8][96];
    __shared__ int   li[8][96];
    __shared__ float sw3[8][3];
    __shared__ int   si3[8][3];

    int b  = blockIdx.x / (N1_ / 8);
    int n0 = (blockIdx.x % (N1_ / 8)) * 8;
    for (int i = threadIdx.x; i < N2_; i += 256) {
        s2x[i] = xyz2[b*3*N2_ + i];
        s2y[i] = xyz2[b*3*N2_ + N2_ + i];
        s2z[i] = xyz2[b*3*N2_ + 2*N2_ + i];
    }
    __syncthreads();

    int wv = threadIdx.x >> 5, lane = threadIdx.x & 31;
    int n = n0 + wv;
    float qx = xyz1[b*3*N1_ + n];
    float qy = xyz1[b*3*N1_ + N1_ + n];
    float qz = xyz1[b*3*N1_ + 2*N1_ + n];

    float d0 = 3e38f, d1 = 3e38f, d2 = 3e38f;
    int   i0 = 0, i1 = 0, i2 = 0;
    for (int j = lane; j < N2_; j += 32) {
        float dx = qx - s2x[j], dy = qy - s2y[j], dz = qz - s2z[j];
        float d = dx*dx + dy*dy + dz*dz;
        if (d < d2) {
            if (d < d0)      { d2=d1; i2=i1; d1=d0; i1=i0; d0=d; i0=j; }
            else if (d < d1) { d2=d1; i2=i1; d1=d;  i1=j; }
            else             { d2=d;  i2=j; }
        }
    }
    ld[wv][lane*3+0] = d0; li[wv][lane*3+0] = i0;
    ld[wv][lane*3+1] = d1; li[wv][lane*3+1] = i1;
    ld[wv][lane*3+2] = d2; li[wv][lane*3+2] = i2;
    __syncthreads();

    if (lane == 0) {
        float m0 = 3e38f, m1 = 3e38f, m2 = 3e38f; int j0 = 0, j1 = 0, j2 = 0;
        for (int e = 0; e < 96; ++e) {
            float d = ld[wv][e]; int ix = li[wv][e];
            if (d < m2) {
                if (d < m0)      { m2=m1; j2=j1; m1=m0; j1=j0; m0=d; j0=ix; }
                else if (d < m1) { m2=m1; j2=j1; m1=d;  j1=ix; }
                else             { m2=d;  j2=ix; }
            }
        }
        float w0 = 1.0f / fmaxf(m0, 1e-10f);
        float w1 = 1.0f / fmaxf(m1, 1e-10f);
        float w2 = 1.0f / fmaxf(m2, 1e-10f);
        float s = w0 + w1 + w2;
        sw3[wv][0] = w0/s; sw3[wv][1] = w1/s; sw3[wv][2] = w2/s;
        si3[wv][0] = j0;   si3[wv][1] = j1;   si3[wv][2] = j2;
    }
    __syncthreads();

    float wA = sw3[wv][0], wB = sw3[wv][1], wC = sw3[wv][2];
    int   jA = si3[wv][0], jB = si3[wv][1], jC = si3[wv][2];
    const float* fb = f2 + (long long)b * C2_ * N2_;
    float* ob = interp + ((long long)b * N1_ + n) * C2_;
    for (int c = lane; c < C2_; c += 32) {
        const float* fc = fb + (long long)c * N2_;
        ob[c] = wA*fc[jA] + wB*fc[jB] + wC*fc[jC];
    }
}

// ---------------------------------------------------------------------------
// kNN (top-16) within xyz1 + Gaussian kernel density
// one wave32 per point: lane-local sorted top-16, LDS merge with shuffle
// min-reduction. knn_idx: [B][N1][K] int; dens: [B][N1] f32
// ---------------------------------------------------------------------------
__global__ __launch_bounds__(256) void knn_density_kernel(
    const float* __restrict__ xyz1, int* __restrict__ knn_idx,
    float* __restrict__ dens)
{
    __shared__ float sd[8][512];
    __shared__ int   si[8][512];

    int b  = blockIdx.x / (N1_ / 8);
    int n0 = (blockIdx.x % (N1_ / 8)) * 8;
    int wv = threadIdx.x >> 5, lane = threadIdx.x & 31;
    int n = n0 + wv;

    const float* xb = xyz1 + (long long)b * 3 * N1_;
    float qx = xb[n], qy = xb[N1_ + n], qz = xb[2*N1_ + n];

    float bd[K_]; int bi[K_];
    #pragma unroll
    for (int r = 0; r < K_; ++r) { bd[r] = 3e38f; bi[r] = 0; }

    float sume = 0.0f;
    for (int j = lane; j < N1_; j += 32) {
        float dx = qx - xb[j], dy = qy - xb[N1_ + j], dz = qz - xb[2*N1_ + j];
        float d = dx*dx + dy*dy + dz*dz;
        sume += __expf(-50.0f * d);              // 1/(2*sigma^2) = 50
        if (d < bd[K_-1]) {
            int pos = K_ - 1;
            while (pos > 0 && bd[pos-1] > d) {
                bd[pos] = bd[pos-1]; bi[pos] = bi[pos-1]; --pos;
            }
            bd[pos] = d; bi[pos] = j;
        }
    }
    #pragma unroll
    for (int r = 0; r < K_; ++r) { sd[wv][lane*K_ + r] = bd[r]; si[wv][lane*K_ + r] = bi[r]; }

    #pragma unroll
    for (int m = 16; m > 0; m >>= 1) sume += __shfl_xor(sume, m, 32);
    if (lane == 0) dens[(long long)b*N1_ + n] = sume * (4.0f / (float)N1_); // /N1 /(2.5*sigma)
    __syncthreads();

    int* outp = knn_idx + ((long long)b*N1_ + n) * K_;
    for (int r = 0; r < K_; ++r) {
        float mv = 3e38f; int mp = 0;
        for (int j = lane; j < 512; j += 32) {
            float v = sd[wv][j];
            if (v < mv) { mv = v; mp = j; }
        }
        #pragma unroll
        for (int m = 16; m > 0; m >>= 1) {
            float ov = __shfl_xor(mv, m, 32);
            int   op = __shfl_xor(mp, m, 32);
            if (ov < mv) { mv = ov; mp = op; }
        }
        if (lane == 0) { outp[r] = si[wv][mp]; sd[wv][mp] = 3e38f; }
        __syncthreads();
    }
}

// ---------------------------------------------------------------------------
// DensityNet: 1 -> 8 -> 8 -> 1, BN+ReLU each layer (bug preserved: ReLU on all)
// ---------------------------------------------------------------------------
__global__ void dnet_kernel(const float* __restrict__ dens, float* __restrict__ ds,
    const float* w0, const float* b0, const float* g0, const float* e0,
    const float* w1, const float* b1, const float* g1, const float* e1,
    const float* w2, const float* b2, const float* g2, const float* e2)
{
    int t = blockIdx.x * 256 + threadIdx.x;
    if (t >= B_ * N1_) return;
    float x = dens[t];
    float h1[8], h2[8];
    #pragma unroll
    for (int o = 0; o < 8; ++o) h1[o] = bnrelu(w0[o]*x + b0[o], g0[o], e0[o]);
    #pragma unroll
    for (int o = 0; o < 8; ++o) {
        float s = b1[o];
        #pragma unroll
        for (int i = 0; i < 8; ++i) s += w1[o*8 + i] * h1[i];
        h2[o] = bnrelu(s, g1[o], e1[o]);
    }
    float s = b2[0];
    #pragma unroll
    for (int i = 0; i < 8; ++i) s += w2[i] * h2[i];
    ds[t] = bnrelu(s, g2[0], e2[0]);
}

// ---------------------------------------------------------------------------
// WeightNet on localized coords: 3 -> 8 -> 8 -> 16
// wts layout: [B][N1][K][16] f32
// ---------------------------------------------------------------------------
__global__ void wnet_kernel(const float* __restrict__ xyz1,
    const int* __restrict__ knn_idx,
    const float* w0, const float* b0, const float* g0, const float* e0,
    const float* w1, const float* b1, const float* g1, const float* e1,
    const float* w2, const float* b2, const float* g2, const float* e2,
    float* __restrict__ wts)
{
    int t = blockIdx.x * 256 + threadIdx.x;
    if (t >= B_ * N1_ * K_) return;
    int n = (t / K_) & (N1_ - 1);
    int b = t / (K_ * N1_);
    const float* xb = xyz1 + (long long)b * 3 * N1_;
    int j = knn_idx[t];
    float gx = xb[j] - xb[n];
    float gy = xb[N1_ + j] - xb[N1_ + n];
    float gz = xb[2*N1_ + j] - xb[2*N1_ + n];
    float h1[8], h2[8];
    #pragma unroll
    for (int o = 0; o < 8; ++o)
        h1[o] = bnrelu(w0[o*3]*gx + w0[o*3+1]*gy + w0[o*3+2]*gz + b0[o], g0[o], e0[o]);
    #pragma unroll
    for (int o = 0; o < 8; ++o) {
        float s = b1[o];
        #pragma unroll
        for (int i = 0; i < 8; ++i) s += w1[o*8 + i] * h1[i];
        h2[o] = bnrelu(s, g1[o], e1[o]);
    }
    float* op = wts + (long long)t * WN_;
    #pragma unroll
    for (int o = 0; o < WN_; ++o) {
        float s = b2[o];
        #pragma unroll
        for (int i = 0; i < 8; ++i) s += w2[o*8 + i] * h2[i];
        op[o] = bnrelu(s, g2[o], e2[o]);
    }
}

// ---------------------------------------------------------------------------
// Density-weighted PointConv aggregation + pack to f16 GEMM layout:
// X0[((b*N1+n)*16 + o)*256 + c] = sum_k interp[idx[k]][c]*ds[idx[k]] * wts[k][o]
// one block (256 thr) per point; K=16 contraction in registers
// ---------------------------------------------------------------------------
__global__ __launch_bounds__(256) void agg_pack_kernel(
    const float* __restrict__ interp, const float* __restrict__ ds,
    const float* __restrict__ wts, const int* __restrict__ knn_idx,
    f16* __restrict__ X0)
{
    __shared__ float sf[K_][C2_ + 1];
    __shared__ float swt[K_][WN_];
    __shared__ int   sk[K_];
    __shared__ float sden[K_];

    int p = blockIdx.x;            // b*N1 + n
    int b = p >> 12;               // / N1_
    int tid = threadIdx.x;
    if (tid < K_) {
        int j = knn_idx[(long long)p * K_ + tid];
        sk[tid]   = j;
        sden[tid] = ds[(long long)b * N1_ + j];
    }
    __syncthreads();
    {   // fold gathered density into the weight tile
        int k = tid >> 4, o = tid & 15;
        swt[k][o] = wts[(long long)p * (K_ * WN_) + tid] * sden[k];
    }
    for (int k = 0; k < K_; ++k)
        sf[k][tid] = interp[((long long)b * N1_ + sk[k]) * C2_ + tid];
    __syncthreads();

    float acc[WN_];
    #pragma unroll
    for (int o = 0; o < WN_; ++o) acc[o] = 0.0f;
    #pragma unroll
    for (int k = 0; k < K_; ++k) {
        float f = sf[k][tid];
        #pragma unroll
        for (int o = 0; o < WN_; ++o) acc[o] += f * swt[k][o];
    }
    f16* out = X0 + ((long long)p * WN_) * C2_ + tid;
    #pragma unroll
    for (int o = 0; o < WN_; ++o) out[(long long)o * C2_] = (f16)acc[o];
}

// ---------------------------------------------------------------------------
// Pack feat_points1 (transposed, f16) into tail of mlp input rows
// ---------------------------------------------------------------------------
__global__ void pack_f1_kernel(const float* __restrict__ f1, f16* __restrict__ X1)
{
    int p = blockIdx.x;            // b*N1 + n
    int c = threadIdx.x;
    int b = p >> 12, n = p & (N1_ - 1);
    X1[(long long)p * FLAT_ + WN_*MLP0_ + c] =
        (f16)f1[((long long)b * C1_ + c) * N1_ + n];
}

// ---------------------------------------------------------------------------
// WMMA fragment loaders (CDNA5 ISA 7.12.2 layouts)
//   A 16x32 f16: lane half selects K {0..7,16..23} vs {8..15,24..31}
//     -> base already includes koffA; halves at +0 and +16
//   B 32x16 f16: lane = N, lane half selects K {0..7,16..23} vs {8..15,24..31}
//     -> base already includes koffB; halves at +0 and +16
// ---------------------------------------------------------------------------
__device__ __forceinline__ v16h ldfragA(const f16* p) {
    union { v16h v; v8h h[2]; } u;
    u.h[0] = *(const v8h*)(p);
    u.h[1] = *(const v8h*)(p + 16);
    return u.v;
}
__device__ __forceinline__ v16h ldfragB(const f16* p) {
    union { v16h v; v8h h[2]; } u;
    u.h[0] = *(const v8h*)(p);
    u.h[1] = *(const v8h*)(p + 16);
    return u.v;
}

// ---------------------------------------------------------------------------
// WMMA GEMM:  Y[m][n] = relu(bn( sum_k X[m][k]*W[n][k] + bias[n] ))
// X: [M][Kd] f16 row-major, W: [N][Kd] f16 row-major (== B transposed).
// Each wave owns a 32x64 output tile (2 M-tiles x 4 N-tiles, 8 v8f accums):
// B fragments are reused across 2 WMMAs, and the K-loop is software-pipelined
// with two register fragment sets so loads for step i+1 are in flight while
// the 8 WMMAs of step i execute (avoids s_wait_loadcnt 0 per WMMA).
// Output index remap: oi = (m>>lgGR)*GS + (m&((1<<lgGR)-1))*RS + n
// ---------------------------------------------------------------------------
__global__ __launch_bounds__(256) void gemm_wmma_kernel(
    const f16* __restrict__ X, const f16* __restrict__ W,
    const float* __restrict__ bias, const float* __restrict__ gamma,
    const float* __restrict__ beta,
    f16* __restrict__ out_h, float* __restrict__ out_f,
    int M, int N, int Kd, int lgGR, long long GS, long long RS)
{
    int wave = threadIdx.x >> 5;
    int lane = threadIdx.x & 31;
    long long wgid = (long long)blockIdx.x * 8 + wave;
    int ntn = N >> 6;                               // 64-wide n-tiles
    long long total = ((long long)M >> 5) * ntn;    // 32-tall m-tiles
    if (wgid >= total) return;
    int       tn = (int)(wgid % ntn) << 6;
    long long tm = (wgid / ntn) << 5;

    int mrow  = lane & 15;
    int nc    = lane & 15;
    int koffA = (lane < 16) ? 0 : 8;
    int koffB = (lane < 16) ? 0 : 8;

    const f16* Ap[2];
    Ap[0] = X + (tm + mrow) * (long long)Kd + koffA;
    Ap[1] = Ap[0] + 16LL * Kd;
    const f16* Bp[4];
    #pragma unroll
    for (int j = 0; j < 4; ++j)
        Bp[j] = W + (long long)(tn + j*16 + nc) * Kd + koffB;

    v8f acc[2][4] = {};
    v16h a0[2], a1[2], b0[4], b1[4];

    // prologue: fragment set 0 @ k=0
    #pragma unroll
    for (int mi = 0; mi < 2; ++mi) a0[mi] = ldfragA(Ap[mi]);
    #pragma unroll
    for (int j = 0; j < 4; ++j)    b0[j]  = ldfragB(Bp[j]);

    for (int k0 = 0; k0 < Kd; k0 += 64) {
        // in-flight: set 1 @ k0+32
        #pragma unroll
        for (int mi = 0; mi < 2; ++mi) a1[mi] = ldfragA(Ap[mi] + k0 + 32);
        #pragma unroll
        for (int j = 0; j < 4; ++j)    b1[j]  = ldfragB(Bp[j] + k0 + 32);
        // compute set 0
        #pragma unroll
        for (int mi = 0; mi < 2; ++mi)
            #pragma unroll
            for (int j = 0; j < 4; ++j)
                acc[mi][j] = __builtin_amdgcn_wmma_f32_16x16x32_f16(
                    false, a0[mi], false, b0[j], (short)0, acc[mi][j], false, false);
        // in-flight: set 0 @ k0+64 (skipped on last iteration)
        if (k0 + 64 < Kd) {
            #pragma unroll
            for (int mi = 0; mi < 2; ++mi) a0[mi] = ldfragA(Ap[mi] + k0 + 64);
            #pragma unroll
            for (int j = 0; j < 4; ++j)    b0[j]  = ldfragB(Bp[j] + k0 + 64);
        }
        // compute set 1
        #pragma unroll
        for (int mi = 0; mi < 2; ++mi)
            #pragma unroll
            for (int j = 0; j < 4; ++j)
                acc[mi][j] = __builtin_amdgcn_wmma_f32_16x16x32_f16(
                    false, a1[mi], false, b1[j], (short)0, acc[mi][j], false, false);
    }

    long long grmask = (1LL << lgGR) - 1;
    #pragma unroll
    for (int j = 0; j < 4; ++j) {
        int ncol = tn + j*16 + nc;
        float bb = bias[ncol];
        float gg = gamma[ncol];
        float ee = beta[ncol];
        #pragma unroll
        for (int mi = 0; mi < 2; ++mi)
            #pragma unroll
            for (int r = 0; r < 8; ++r) {
                long long m = tm + mi*16 + r + ((lane >> 4) << 3);
                float y = bnrelu(acc[mi][j][r] + bb, gg, ee);
                long long oi = ((m >> lgGR) * GS) + ((m & grmask) * RS) + ncol;
                if (out_h) out_h[oi] = (f16)y;
                else       out_f[oi] = y;
            }
    }
}

// ---------------------------------------------------------------------------
// Host launcher
// Input order: setup_inputs() dict order (xyz1, xyz2, feat_points1,
// feat_points2, params), params flattened jax-pytree style (sorted keys:
// conv0, dnet, mlp, wnet; each layer tuple (w, b, g, be)).
// ---------------------------------------------------------------------------
extern "C" void kernel_launch(void* const* d_in, const int* in_sizes, int n_in,
                              void* d_out, int out_size, void* d_ws, size_t ws_size,
                              hipStream_t stream)
{
    (void)in_sizes; (void)n_in; (void)out_size; (void)ws_size;

    const float* xyz1 = (const float*)d_in[0];
    const float* xyz2 = (const float*)d_in[1];
    const float* f1   = (const float*)d_in[2];
    const float* f2   = (const float*)d_in[3];
    // conv0 (w,b,g,be)
    const float* c0w = (const float*)d_in[4];
    const float* c0b = (const float*)d_in[5];
    const float* c0g = (const float*)d_in[6];
    const float* c0e = (const float*)d_in[7];
    // dnet: 3 layers x (w,b,g,be) -> indices 8..19
    const float* dn[12]; for (int i = 0; i < 12; ++i) dn[i] = (const float*)d_in[8 + i];
    // mlp: 2 layers -> 20..27
    const float* m0w = (const float*)d_in[20];
    const float* m0b = (const float*)d_in[21];
    const float* m0g = (const float*)d_in[22];
    const float* m0e = (const float*)d_in[23];
    const float* m1w = (const float*)d_in[24];
    const float* m1b = (const float*)d_in[25];
    const float* m1g = (const float*)d_in[26];
    const float* m1e = (const float*)d_in[27];
    // wnet: 3 layers -> 28..39
    const float* wn[12]; for (int i = 0; i < 12; ++i) wn[i] = (const float*)d_in[28 + i];

    // workspace carve (256B aligned)
    uintptr_t p = (uintptr_t)d_ws;
    auto take = [&](size_t bytes) -> void* {
        void* r = (void*)p;
        p = (p + bytes + 255) & ~(uintptr_t)255;
        return r;
    };
    float* interp = (float*)take((size_t)B_*N1_*C2_*4);        // 16.8 MB
    int*   kidx   = (int*)  take((size_t)B_*N1_*K_*4);         //  1.0 MB
    float* dens   = (float*)take((size_t)B_*N1_*4);
    float* dsv    = (float*)take((size_t)B_*N1_*4);
    float* wtsv   = (float*)take((size_t)B_*N1_*K_*WN_*4);     // 16.8 MB
    f16*   X0     = (f16*)  take((size_t)B_*N1_*WN_*C2_*2);    // 134 MB
    f16*   X1     = (f16*)  take((size_t)B_*N1_*FLAT_*2);      // 143 MB
    f16*   X2     = (f16*)  take((size_t)B_*N1_*MLP0_*2);      //   8 MB
    f16*   w0h    = (f16*)  take((size_t)MLP0_*C2_*2);
    f16*   w1h    = (f16*)  take((size_t)MLP0_*FLAT_*2);
    f16*   w2h    = (f16*)  take((size_t)MLP1_*MLP0_*2);

    // weight conversion f32 -> f16
    cvt_f32_f16_kernel<<<(MLP0_*C2_ + 255)/256, 256, 0, stream>>>(c0w, w0h, MLP0_*C2_);
    cvt_f32_f16_kernel<<<(MLP0_*FLAT_ + 255)/256, 256, 0, stream>>>(m0w, w1h, MLP0_*FLAT_);
    cvt_f32_f16_kernel<<<(MLP1_*MLP0_ + 255)/256, 256, 0, stream>>>(m1w, w2h, MLP1_*MLP0_);

    // geometry / gather stages
    three_nn_interp_kernel<<<B_*N1_/8, 256, 0, stream>>>(xyz1, xyz2, f2, interp);
    knn_density_kernel<<<B_*N1_/8, 256, 0, stream>>>(xyz1, kidx, dens);
    dnet_kernel<<<(B_*N1_)/256, 256, 0, stream>>>(dens, dsv,
        dn[0], dn[1], dn[2], dn[3], dn[4], dn[5], dn[6], dn[7], dn[8], dn[9], dn[10], dn[11]);
    wnet_kernel<<<(B_*N1_*K_)/256, 256, 0, stream>>>(xyz1, kidx,
        wn[0], wn[1], wn[2], wn[3], wn[4], wn[5], wn[6], wn[7], wn[8], wn[9], wn[10], wn[11],
        wtsv);
    agg_pack_kernel<<<B_*N1_, 256, 0, stream>>>(interp, dsv, wtsv, kidx, X0);
    pack_f1_kernel<<<B_*N1_, 256, 0, stream>>>(f1, X1);

    // conv0: [262144 x 256] @ [256 x 256]^T -> scattered into X1 rows
    // waves = (M/32)*(N/64) = 32768 -> 4096 blocks
    gemm_wmma_kernel<<<4096, 256, 0, stream>>>(X0, w0h, c0b, c0g, c0e,
        X1, nullptr, B_*N1_*WN_, MLP0_, C2_, 4, FLAT_, 256);

    // mlp layer 1: [16384 x 4352] @ [4352 x 256]^T -> X2   (2048 waves -> 256 blocks)
    gemm_wmma_kernel<<<256, 256, 0, stream>>>(X1, w1h, m0b, m0g, m0e,
        X2, nullptr, B_*N1_, MLP0_, FLAT_, 0, MLP0_, 0);

    // mlp layer 2: [16384 x 256] @ [256 x 256]^T -> d_out (f32)
    gemm_wmma_kernel<<<256, 256, 0, stream>>>(X2, w2h, m1b, m1g, m1e,
        nullptr, (float*)d_out, B_*N1_, MLP1_, MLP0_, 0, MLP1_, 0);
}
